// GaitnetCritic_15212774162768
// MI455X (gfx1250) — matrix operand order repaired and checked by reading, not compiled, not run on hardware
//
#include <hip/hip_runtime.h>

typedef __attribute__((ext_vector_type(16))) _Float16 v16h;
typedef __attribute__((ext_vector_type(8)))  _Float16 v8h;
typedef __attribute__((ext_vector_type(8)))  float    v8f;

#define BATCH 16384
#define ROWS_U (BATCH * 16)   // 262144 unique/trunk rows

#if __has_builtin(__builtin_amdgcn_tensor_load_to_lds) && __has_builtin(__builtin_amdgcn_s_wait_tensorcnt)
#define HAVE_TDM 1
typedef __attribute__((ext_vector_type(4))) unsigned int u32x4;
typedef __attribute__((ext_vector_type(8))) int          i32x8;
typedef __attribute__((ext_vector_type(4))) int          i32x4;
#else
#define HAVE_TDM 0
#endif

#if __has_builtin(__builtin_amdgcn_sched_barrier)
#define SCHED_PIN() __builtin_amdgcn_sched_barrier(0)
#else
#define SCHED_PIN()
#endif

// ---------------------------------------------------------------------------
// Weight transpose + f16 convert: W (K x N, row-major f32) -> Wt (Npad x Kpad f16)
// ---------------------------------------------------------------------------
__global__ __launch_bounds__(256) void transpose_w_f16(
    const float* __restrict__ W, _Float16* __restrict__ Wt,
    int K, int N, int Kpad, int Npad) {
  int idx = blockIdx.x * 256 + threadIdx.x;
  int tot = Kpad * Npad;
  if (idx >= tot) return;
  int n = idx / Kpad;
  int k = idx - n * Kpad;
  _Float16 v = (_Float16)0.0f;
  if (k < K && n < N) v = (_Float16)W[k * N + n];
  Wt[idx] = v;
}

// obs[:, :22] -> As (B x 32 f16, zero padded)
__global__ __launch_bounds__(256) void prep_shared_in(
    const float* __restrict__ obs, _Float16* __restrict__ As) {
  int idx = blockIdx.x * 256 + threadIdx.x;   // B*32 threads
  int env = idx >> 5, c = idx & 31;
  _Float16 v = (_Float16)0.0f;
  if (c < 22) v = (_Float16)obs[env * 150 + c];
  As[idx] = v;
}

// obs[:, 22:].reshape(B,16,8) -> Au (B*16 x 32 f16, zero padded) + no-op mask
__global__ __launch_bounds__(256) void prep_unique_in(
    const float* __restrict__ obs, _Float16* __restrict__ Au,
    unsigned char* __restrict__ mask) {
  int idx = blockIdx.x * 256 + threadIdx.x;   // B*16*32 threads
  int row = idx >> 5, c = idx & 31;
  int env = row >> 4, u = row & 15;
  const float* src = obs + env * 150 + 22 + u * 8;
  _Float16 v = (_Float16)0.0f;
  if (c < 8) v = (_Float16)src[c];
  Au[idx] = v;
  if (c == 0) {
    mask[row] = (src[4] == 1.0f) || (src[7] >= 2.0f) ? 1 : 0;
  }
}

// trunk_in = concat(shared_emb[env] (128), mask ? no_op : enc[row] (64)) -> 192 f16
__global__ __launch_bounds__(256) void build_trunk_in(
    const _Float16* __restrict__ Sh, const _Float16* __restrict__ Enc,
    const unsigned char* __restrict__ mask, const float* __restrict__ noop,
    _Float16* __restrict__ Tin) {
  int idx = blockIdx.x * 256 + threadIdx.x;   // B*16*48 threads, 4 cols each
  int row = idx / 48;
  int c4  = (idx - row * 48) * 4;
  int env = row >> 4;
  _Float16 o0, o1, o2, o3;
  if (c4 < 128) {
    const _Float16* s = Sh + (size_t)env * 128 + c4;
    o0 = s[0]; o1 = s[1]; o2 = s[2]; o3 = s[3];
  } else {
    int cc = c4 - 128;
    if (mask[row]) {
      o0 = (_Float16)noop[cc + 0]; o1 = (_Float16)noop[cc + 1];
      o2 = (_Float16)noop[cc + 2]; o3 = (_Float16)noop[cc + 3];
    } else {
      const _Float16* e = Enc + (size_t)row * 64 + cc;
      o0 = e[0]; o1 = e[1]; o2 = e[2]; o3 = e[3];
    }
  }
  _Float16* d = Tin + (size_t)row * 192 + c4;
  d[0] = o0; d[1] = o1; d[2] = o2; d[3] = o3;
}

#if HAVE_TDM
// ---------------------------------------------------------------------------
// Tensor Data Mover: DMA the whole transposed weight matrix (NROWS x Kpad f16,
// row-major, contiguous) into LDS offset 0, inserting 4 dwords (8 halves) of
// padding after every row so the LDS image matches LROW = Kpad + 8.
// D# packed per CDNA5 ISA. Issued by one wave.
// This toolchain declares the 6-arg builtin form:
//   (u32x4 g0, i32x8 g1, i32x4 g2, i32x4 g3, i32x8 extra, i32 cpol)
// ---------------------------------------------------------------------------
template <int NROWS, int Kpad>
__device__ inline void tdm_load_weights(const _Float16* Wt) {
  constexpr int rowdw = Kpad / 2;                       // dwords per row: 16/64/128
  constexpr int ci = (rowdw == 16) ? 3 : (rowdw == 64) ? 5 : 6;  // 2^(ci+1) dwords
  unsigned long long ga = (unsigned long long)(__SIZE_TYPE__)Wt;
  u32x4 g0;
  g0[0] = 1u;                                           // count=1, user mode
  g0[1] = 0u;                                           // lds_addr = 0
  g0[2] = (unsigned int)(ga & 0xFFFFFFFFu);             // global_addr[31:0]
  g0[3] = (unsigned int)(((ga >> 32) & 0x01FFFFFFu)     // global_addr[56:32]
                         | 0x80000000u);                // type=2 ("image")
  i32x8 g1;
  g1[0] = (1 << 16)      // data_size = 2 bytes
        | (1 << 20)      // pad_enable
        | (ci << 22)     // pad_interval: every row
        | (3 << 25);     // pad_amount: 4 dwords (8 halves)
  g1[1] = (int)((Kpad & 0xFFFF) << 16);                 // tensor_dim0[15:0]
  g1[2] = (int)(((Kpad >> 16) & 0xFFFF) | ((NROWS & 0xFFFF) << 16)); // td0 hi | td1 lo
  g1[3] = (int)(((NROWS >> 16) & 0xFFFF) | ((Kpad & 0xFFFF) << 16)); // td1 hi | tile_dim0
  g1[4] = (int)(NROWS & 0xFFFF);                        // tile_dim1 | tile_dim2=0
  g1[5] = Kpad;                                         // tensor_dim0_stride[31:0]
  g1[6] = 0;                                            // stride hi | td1_stride lo
  g1[7] = 0;
  i32x4 g2; g2[0] = 1; g2[1] = 1; g2[2] = 0; g2[3] = 0; // dim2/dim3 = 1 (unused)
  i32x4 g3; g3[0] = 0; g3[1] = 0; g3[2] = 0; g3[3] = 0;
  i32x8 g4;                                             // extra group (zero-filled)
  g4[0] = 0; g4[1] = 0; g4[2] = 0; g4[3] = 0;
  g4[4] = 0; g4[5] = 0; g4[6] = 0; g4[7] = 0;
  __builtin_amdgcn_tensor_load_to_lds(g0, g1, g2, g3, g4, 0);
  __builtin_amdgcn_s_wait_tensorcnt(0);
}
#endif

// ---------------------------------------------------------------------------
// Generic fused GEMM + bias + (relu) layer via v_wmma_f32_16x16x32_f16.
//   A  : M x Kpad f16 row-major (Kpad = KT*32), streamed from global (L2),
//        double-buffered one K-step ahead.
//   Wt : (NT*16) x Kpad f16 transposed weights; staged once per block into
//        LDS (TDM DMA where row-dwords is pow2, else cooperative copy) with
//        +8-half row padding -> conflict-free ds_load_b128 B reads.
//   B operand kept two tiles in flight; sched_barrier pins the prefetch
//   above the consuming WMMA so the allocator cannot collapse the buffers.
// 256 threads = 8 waves; each wave computes a 16-row stripe across all N.
// ---------------------------------------------------------------------------
template <int NT, int KT, int RELU>
__global__ __launch_bounds__(256) void wmma_mlp_layer(
    const _Float16* __restrict__ A,
    const _Float16* __restrict__ Wt,
    const float* __restrict__ bias,
    _Float16* __restrict__ outH, int ldoH,
    float* __restrict__ outF, int ldoF,
    int M, int N) {
  constexpr int Kpad  = KT * 32;
  constexpr int NROWS = NT * 16;
  constexpr int LROW  = Kpad + 8;   // halves; stride dwords = Kpad/2+4 (bank-swizzled)
  extern __shared__ _Float16 wlds[];

#if HAVE_TDM
  constexpr bool kTdm = (Kpad == 32 || Kpad == 128 || Kpad == 256);
#else
  constexpr bool kTdm = false;
#endif
  if constexpr (kTdm) {
#if HAVE_TDM
    if (threadIdx.x < 32) tdm_load_weights<NROWS, Kpad>(Wt);  // one wave issues DMA
#endif
  } else {
    // cooperative staging: global b128 loads -> ds_store_b128
    constexpr int CH = (NROWS * Kpad) / 8;   // 16-byte chunks (multiple of 256)
    for (int c = threadIdx.x; c < CH; c += 256) {
      int r  = c / (Kpad / 8);
      int kk = (c - r * (Kpad / 8)) * 8;
      v8h w = *(const v8h*)(Wt + (size_t)r * Kpad + kk);
      *(v8h*)(&wlds[r * LROW + kk]) = w;
    }
  }
  __syncthreads();

  const int lane = threadIdx.x & 31;
  const int wave = threadIdx.x >> 5;
  const int rowbase = blockIdx.x * 128 + wave * 16;
  if (rowbase >= M) return;            // uniform per wave: EXEC stays all-ones

  const int n0  = lane & 15;           // N index (B/C/D layouts) and M index (A layout)
  const int hiA = (lane >> 4) << 3;    // A K-offset: 0 or 8
  const int hiB = (lane >> 4) << 3;    // B K-offset: 0 or 8
  const int mhi = (lane >> 4) << 3;    // D row offset: 0 or 8

  v8f acc[NT];
#pragma unroll
  for (int t = 0; t < NT; ++t) {
    int n = t * 16 + n0;
    float bv = (n < N) ? bias[n] : 0.0f;
    v8f c = {bv, bv, bv, bv, bv, bv, bv, bv};
    acc[t] = c;
  }

  // B operand from LDS: lane holds N = lane%16, K = k0 + hiB + 0..7 and k0 + 16 + hiB + 0..7
  const _Float16* lbase = &wlds[(size_t)n0 * LROW + hiB];
  auto ldB = [&](int kb, int t) {
    const _Float16* p = lbase + (size_t)t * 16 * LROW + kb * 32;
    union { v16h v; v8h h[2]; } u;
    u.h[0] = *(const v8h*)(p);
    u.h[1] = *(const v8h*)(p + 16);
    return u.v;
  };

  // A operand: ISA 16-bit A 16x32 layout. Lane holds M = lane%16;
  // halves 0..7 -> K = k0 + hiA + 0..7 ; halves 8..15 -> K = k0 + 16 + hiA + 0..7
  const _Float16* arow = A + (size_t)(rowbase + n0) * Kpad;
  v8h a0n = *(const v8h*)(arow + hiA);
  v8h a1n = *(const v8h*)(arow + 16 + hiA);

  constexpr int TOT = KT * NT;
  v16h b0 = ldB(0, 0);
  v16h b1 = (TOT > 1) ? ldB(1 / NT, 1 % NT) : b0;

  union { v16h v; v8h h[2]; } au;
#pragma unroll
  for (int i = 0; i < TOT; ++i) {
    const int kb = i / NT;
    const int t  = i % NT;
    if (t == 0) {
      au.h[0] = a0n;
      au.h[1] = a1n;
      if (kb + 1 < KT) {                     // prefetch next A K-step (global)
        a0n = *(const v8h*)(arow + (kb + 1) * 32 + hiA);
        a1n = *(const v8h*)(arow + (kb + 1) * 32 + 16 + hiA);
      }
    }
    v16h b = b0;
    b0 = b1;
    if (i + 2 < TOT) b1 = ldB((i + 2) / NT, (i + 2) % NT);   // keep 2 B tiles in flight
    SCHED_PIN();   // pin prefetch above the WMMA: forces distinct buffers
    acc[t] = __builtin_amdgcn_wmma_f32_16x16x32_f16(
        /*neg_a=*/false, au.v, /*neg_b=*/false, b,
        /*c_mod=*/(short)0, acc[t], /*reuse_a=*/false, /*reuse_b=*/false);
  }

  // D layout: VGPR r -> row = rowbase + mhi + r, col = lane%16
#pragma unroll
  for (int t = 0; t < NT; ++t) {
    int n = t * 16 + n0;
    if (n < N) {
#pragma unroll
      for (int r = 0; r < 8; ++r) {
        float v = acc[t][r];
        if (RELU) v = fmaxf(v, 0.0f);
        size_t row = (size_t)(rowbase + mhi + r);
        if (outH) outH[row * (size_t)ldoH + n] = (_Float16)v;
        if (outF) outF[row * (size_t)ldoF + n] = v;
      }
    }
  }
}

// ---------------------------------------------------------------------------
// Tiny comb head: per-env f32 MLP 16 -> 32 -> 32 -> 1 (no final relu)
// ---------------------------------------------------------------------------
__global__ __launch_bounds__(256) void comb_kernel(
    const float* __restrict__ values,
    const float* __restrict__ W1, const float* __restrict__ b1,
    const float* __restrict__ W2, const float* __restrict__ b2,
    const float* __restrict__ W3, const float* __restrict__ b3,
    float* __restrict__ out, int B) {
  int env = blockIdx.x * 256 + threadIdx.x;
  if (env >= B) return;
  float x[16];
#pragma unroll
  for (int i = 0; i < 16; ++i) x[i] = values[(size_t)env * 16 + i];
  float h1[32];
#pragma unroll 4
  for (int j = 0; j < 32; ++j) {
    float s = b1[j];
#pragma unroll
    for (int i = 0; i < 16; ++i) s = fmaf(x[i], W1[i * 32 + j], s);
    h1[j] = fmaxf(s, 0.0f);
  }
  float h2[32];
#pragma unroll 4
  for (int j = 0; j < 32; ++j) {
    float s = b2[j];
#pragma unroll
    for (int i = 0; i < 32; ++i) s = fmaf(h1[i], W2[i * 32 + j], s);
    h2[j] = fmaxf(s, 0.0f);
  }
  float s = b3[0];
#pragma unroll
  for (int i = 0; i < 32; ++i) s = fmaf(h2[i], W3[i], s);
  out[env] = s;
}

// ---------------------------------------------------------------------------
extern "C" void kernel_launch(void* const* d_in, const int* in_sizes, int n_in,
                              void* d_out, int out_size, void* d_ws, size_t ws_size,
                              hipStream_t stream) {
  (void)in_sizes; (void)n_in; (void)out_size;
  const float* obs  = (const float*)d_in[0];
  const float* sW[3] = {(const float*)d_in[1], (const float*)d_in[2], (const float*)d_in[3]};
  const float* sB[3] = {(const float*)d_in[4], (const float*)d_in[5], (const float*)d_in[6]};
  const float* uW[3] = {(const float*)d_in[7], (const float*)d_in[8], (const float*)d_in[9]};
  const float* uB[3] = {(const float*)d_in[10], (const float*)d_in[11], (const float*)d_in[12]};
  const float* noop  = (const float*)d_in[13];
  const float* tW[3] = {(const float*)d_in[14], (const float*)d_in[15], (const float*)d_in[16]};
  const float* tB[3] = {(const float*)d_in[17], (const float*)d_in[18], (const float*)d_in[19]};
  const float* cW[3] = {(const float*)d_in[20], (const float*)d_in[21], (const float*)d_in[22]};
  const float* cB[3] = {(const float*)d_in[23], (const float*)d_in[24], (const float*)d_in[25]};

  char* ws = (char*)d_ws;
  const size_t MB = 1ull << 20;

  // ---- workspace layout (lifetime-aliased; peak ~326 MB) ----
  const size_t WT_S0 = 0;         // 256x32  f16
  const size_t WT_S1 = 16384;     // 256x256 f16
  const size_t WT_S2 = 147456;    // 128x256 f16
  const size_t WT_U0 = 212992;    // 128x32  f16
  const size_t WT_U1 = 221184;    // 128x128 f16
  const size_t WT_U2 = 253952;    // 64x128  f16
  const size_t WT_T0 = 270336;    // 256x192 f16
  const size_t WT_T1 = 368640;    // 128x256 f16
  const size_t WT_T2 = 434176;    // 16x128  f16
  const size_t MASK_O = 438272;          // B*16 u8
  const size_t SH_O   = 700416;          // B x 128 f16 (4 MB)
  const size_t VAL_O  = SH_O + 4 * MB;   // B*16 f32 (1 MB)
  const size_t SEG1   = 6 * MB;          // 128 MB: As/H1/H2/U1, later T1
  const size_t AS_O   = SEG1;                 // B x 32 f16 (1 MB)
  const size_t H1_O   = SEG1 + 1 * MB;        // B x 256 f16 (8 MB)
  const size_t H2_O   = SEG1 + 9 * MB;        // B x 256 f16 (8 MB)
  const size_t U1_O   = SEG1 + 17 * MB;       // B*16 x 128 f16 (64 MB)
  const size_t T1_O   = SEG1;                 // B*16 x 256 f16 (128 MB)
  const size_t SEG2   = SEG1 + 128 * MB; // 96 MB: Au, later Tin
  const size_t AU_O   = SEG2;                 // B*16 x 32 f16 (16 MB)
  const size_t TIN_O  = SEG2;                 // B*16 x 192 f16 (96 MB)
  const size_t SEG3   = SEG2 + 96 * MB;  // 64 MB: U2, later T2
  const size_t U2_O   = SEG3;                 // B*16 x 128 f16 (64 MB)
  const size_t T2_O   = SEG3;                 // B*16 x 128 f16 (64 MB)
  const size_t ENC_O  = SEG3 + 64 * MB;  // B*16 x 64 f16 (32 MB)
  const size_t WS_NEEDED = ENC_O + 32 * MB;
  if (ws_size < WS_NEEDED) return;

  // ---- 1) weight transposes (f32 -> transposed, padded f16) ----
  struct TW { const float* W; size_t off; int K, N, Kpad, Npad; };
  const TW tws[9] = {
      {sW[0], WT_S0,  22, 256,  32, 256}, {sW[1], WT_S1, 256, 256, 256, 256},
      {sW[2], WT_S2, 256, 128, 256, 128},
      {uW[0], WT_U0,   8, 128,  32, 128}, {uW[1], WT_U1, 128, 128, 128, 128},
      {uW[2], WT_U2, 128,  64, 128,  64},
      {tW[0], WT_T0, 192, 256, 192, 256}, {tW[1], WT_T1, 256, 128, 256, 128},
      {tW[2], WT_T2, 128,   1, 128,  16},
  };
  for (int i = 0; i < 9; ++i) {
    int tot = tws[i].Kpad * tws[i].Npad;
    transpose_w_f16<<<(tot + 255) / 256, 256, 0, stream>>>(
        tws[i].W, (_Float16*)(ws + tws[i].off),
        tws[i].K, tws[i].N, tws[i].Kpad, tws[i].Npad);
  }

  // ---- 2) input prep ----
  prep_shared_in<<<(BATCH * 32) / 256, 256, 0, stream>>>(obs, (_Float16*)(ws + AS_O));
  prep_unique_in<<<(ROWS_U * 32) / 256, 256, 0, stream>>>(
      obs, (_Float16*)(ws + AU_O), (unsigned char*)(ws + MASK_O));

  const int gridS = BATCH / 128;   // 128 blocks
  const int gridU = ROWS_U / 128;  // 2048 blocks

  // dynamic LDS bytes per instantiation: NT*16 rows * (KT*32+8) halves * 2B
  auto shb = [](int NT, int KT) { return (size_t)(NT * 16) * (KT * 32 + 8) * 2; };
  (void)hipFuncSetAttribute((const void*)&wmma_mlp_layer<16, 1, 1>, hipFuncAttributeMaxDynamicSharedMemorySize, (int)shb(16, 1));
  (void)hipFuncSetAttribute((const void*)&wmma_mlp_layer<16, 8, 1>, hipFuncAttributeMaxDynamicSharedMemorySize, (int)shb(16, 8));
  (void)hipFuncSetAttribute((const void*)&wmma_mlp_layer<8, 8, 1>,  hipFuncAttributeMaxDynamicSharedMemorySize, (int)shb(8, 8));
  (void)hipFuncSetAttribute((const void*)&wmma_mlp_layer<8, 1, 1>,  hipFuncAttributeMaxDynamicSharedMemorySize, (int)shb(8, 1));
  (void)hipFuncSetAttribute((const void*)&wmma_mlp_layer<8, 4, 1>,  hipFuncAttributeMaxDynamicSharedMemorySize, (int)shb(8, 4));
  (void)hipFuncSetAttribute((const void*)&wmma_mlp_layer<4, 4, 1>,  hipFuncAttributeMaxDynamicSharedMemorySize, (int)shb(4, 4));
  (void)hipFuncSetAttribute((const void*)&wmma_mlp_layer<16, 6, 1>, hipFuncAttributeMaxDynamicSharedMemorySize, (int)shb(16, 6));
  (void)hipFuncSetAttribute((const void*)&wmma_mlp_layer<1, 4, 0>,  hipFuncAttributeMaxDynamicSharedMemorySize, (int)shb(1, 4));

  // ---- 3) shared MLP: 22->256->256->128 (all relu) ----
  wmma_mlp_layer<16, 1, 1><<<gridS, 256, shb(16, 1), stream>>>(
      (const _Float16*)(ws + AS_O), (const _Float16*)(ws + WT_S0), sB[0],
      (_Float16*)(ws + H1_O), 256, nullptr, 0, BATCH, 256);
  wmma_mlp_layer<16, 8, 1><<<gridS, 256, shb(16, 8), stream>>>(
      (const _Float16*)(ws + H1_O), (const _Float16*)(ws + WT_S1), sB[1],
      (_Float16*)(ws + H2_O), 256, nullptr, 0, BATCH, 256);
  wmma_mlp_layer<8, 8, 1><<<gridS, 256, shb(8, 8), stream>>>(
      (const _Float16*)(ws + H2_O), (const _Float16*)(ws + WT_S2), sB[2],
      (_Float16*)(ws + SH_O), 128, nullptr, 0, BATCH, 128);

  // ---- 4) unique MLP: 8->128->128->64 (all relu) ----
  wmma_mlp_layer<8, 1, 1><<<gridU, 256, shb(8, 1), stream>>>(
      (const _Float16*)(ws + AU_O), (const _Float16*)(ws + WT_U0), uB[0],
      (_Float16*)(ws + U1_O), 128, nullptr, 0, ROWS_U, 128);
  wmma_mlp_layer<8, 4, 1><<<gridU, 256, shb(8, 4), stream>>>(
      (const _Float16*)(ws + U1_O), (const _Float16*)(ws + WT_U1), uB[1],
      (_Float16*)(ws + U2_O), 128, nullptr, 0, ROWS_U, 128);
  wmma_mlp_layer<4, 4, 1><<<gridU, 256, shb(4, 4), stream>>>(
      (const _Float16*)(ws + U2_O), (const _Float16*)(ws + WT_U2), uB[2],
      (_Float16*)(ws + ENC_O), 64, nullptr, 0, ROWS_U, 64);

  // ---- 5) mask + concat -> trunk input (B*16 x 192) ----
  build_trunk_in<<<(ROWS_U * 48) / 256, 256, 0, stream>>>(
      (const _Float16*)(ws + SH_O), (const _Float16*)(ws + ENC_O),
      (const unsigned char*)(ws + MASK_O), noop, (_Float16*)(ws + TIN_O));

  // ---- 6) trunk MLP: 192->256->128->1 (final no relu, f32 values out) ----
  wmma_mlp_layer<16, 6, 1><<<gridU, 256, shb(16, 6), stream>>>(
      (const _Float16*)(ws + TIN_O), (const _Float16*)(ws + WT_T0), tB[0],
      (_Float16*)(ws + T1_O), 256, nullptr, 0, ROWS_U, 256);
  wmma_mlp_layer<8, 8, 1><<<gridU, 256, shb(8, 8), stream>>>(
      (const _Float16*)(ws + T1_O), (const _Float16*)(ws + WT_T1), tB[1],
      (_Float16*)(ws + T2_O), 128, nullptr, 0, ROWS_U, 128);
  wmma_mlp_layer<1, 4, 0><<<gridU, 256, shb(1, 4), stream>>>(
      (const _Float16*)(ws + T2_O), (const _Float16*)(ws + WT_T2), tB[2],
      nullptr, 0, (float*)(ws + VAL_O), 1, ROWS_U, 1);

  // ---- 7) comb head: (B,16) -> (B,1) ----
  comb_kernel<<<BATCH / 256, 256, 0, stream>>>(
      (const float*)(ws + VAL_O), cW[0], cB[0], cW[1], cB[1], cW[2], cB[2],
      (float*)d_out, BATCH);
}
